// M10bTranslationAdapter_32899449487368
// MI455X (gfx1250) — matrix operand order repaired and checked by reading, not compiled, not run on hardware
//
#include <hip/hip_runtime.h>
#include <hip/hip_bf16.h>

typedef __attribute__((ext_vector_type(16))) __bf16 v16bf;
typedef __attribute__((ext_vector_type(8)))  float  v8f;
typedef __attribute__((ext_vector_type(4)))  int    v4i;
typedef unsigned short ushort_t;

#define HDIM 2048      // H
#define BATCH 4
#define LSEQ 4096
#define SSEQ 10
#define BL (BATCH * LSEQ)   // 16384 query rows
#define BS (BATCH * SSEQ)   // 40 key/value rows

#define BM 64
#define BN 256
#define BK 32               // one bf16-WMMA K-step per staged tile
#define LDSTR (BK + 8)      // LDS row stride in halfwords: 80B -> 16B aligned, bank-spread

// ---- bf16 split helpers (manual RNE, no __bf16 scalar ops) -----------------
__device__ inline ushort_t bf16_rne(float f) {
  unsigned int u = __float_as_uint(f);
  u += 0x7FFFu + ((u >> 16) & 1u);
  return (ushort_t)(u >> 16);
}
__device__ inline float bf16_f(ushort_t h) {
  return __uint_as_float(((unsigned int)h) << 16);
}
__device__ inline void split_f32(float x, ushort_t& h, ushort_t& l) {
  h = bf16_rne(x);
  l = bf16_rne(x - bf16_f(h));
}
// Build a 16-element bf16 operand from two 16B-aligned LDS segments.
__device__ inline v16bf ld16(const ushort_t* p0, const ushort_t* p1) {
  union { uint4 q[2]; v16bf v; } u;
  u.q[0] = *(const uint4*)p0;
  u.q[1] = *(const uint4*)p1;
  return u.v;
}

// ---- async global->LDS copy (CDNA5), with safe fallback --------------------
#if __has_builtin(__builtin_amdgcn_global_load_async_to_lds_b128)
#define ASYNC_COPY 1
#endif

__device__ inline void cp16(const ushort_t* g, ushort_t* l) {
#ifdef ASYNC_COPY
  __builtin_amdgcn_global_load_async_to_lds_b128(
      (__attribute__((address_space(1))) v4i*)(void*)const_cast<ushort_t*>(g),
      (__attribute__((address_space(3))) v4i*)(void*)l,
      0, 0);
#else
  *(uint4*)l = *(const uint4*)g;
#endif
}
__device__ inline void cp_fence() {
#ifdef ASYNC_COPY
#if __has_builtin(__builtin_amdgcn_s_wait_asynccnt)
  __builtin_amdgcn_s_wait_asynccnt(0);
#else
  asm volatile("s_wait_asynccnt 0x0" ::: "memory");
#endif
#endif
}

// ---------------------------------------------------------------------------
// Pre-pass: split f32 tensor into hi/lo bf16 planes (one pass, no redundancy).
// ---------------------------------------------------------------------------
__global__ __launch_bounds__(256)
void split4_kernel(const float* __restrict__ src, ushort_t* __restrict__ hi,
                   ushort_t* __restrict__ lo, long n4) {
  const long stride = (long)gridDim.x * blockDim.x;
  for (long i = (long)blockIdx.x * blockDim.x + threadIdx.x; i < n4; i += stride) {
    const float4 v = ((const float4*)src)[i];
    ushort_t h0, h1, h2, h3, l0, l1, l2, l3;
    split_f32(v.x, h0, l0); split_f32(v.y, h1, l1);
    split_f32(v.z, h2, l2); split_f32(v.w, h3, l3);
    uint2 hp, lp;
    hp.x = (unsigned)h0 | ((unsigned)h1 << 16);
    hp.y = (unsigned)h2 | ((unsigned)h3 << 16);
    lp.x = (unsigned)l0 | ((unsigned)l1 << 16);
    lp.y = (unsigned)l2 | ((unsigned)l3 << 16);
    ((uint2*)hi)[i] = hp;
    ((uint2*)lo)[i] = lp;
  }
}

// ---------------------------------------------------------------------------
// f32-accurate GEMM on pre-split bf16 planes:
//   C[M, 2048] = A[M, 2048] @ W[2048, 2048]^T
//   A*W ~= Ahi*Whi + Alo*Whi + Ahi*Wlo  (f32 accum in v_wmma_f32_16x16x32_bf16)
// Optional epilogue: C = resid + alpha * acc
// Block 64x256, 8 waves as 2(M) x 4(N); wave tile 32x64 -> 24 WMMA per K-step.
// ---------------------------------------------------------------------------
__global__ __launch_bounds__(256)
void gemm_bf16split_wmma(const ushort_t* __restrict__ Ahg, const ushort_t* __restrict__ Alg,
                         const ushort_t* __restrict__ Whg, const ushort_t* __restrict__ Wlg,
                         float* __restrict__ C, int M,
                         const float* __restrict__ resid,
                         const float* __restrict__ alpha_ptr) {
  __shared__ __attribute__((aligned(16))) ushort_t Ah[BM][LDSTR];
  __shared__ __attribute__((aligned(16))) ushort_t Al[BM][LDSTR];
  __shared__ __attribute__((aligned(16))) ushort_t Wh[BN][LDSTR];
  __shared__ __attribute__((aligned(16))) ushort_t Wl[BN][LDSTR];

  const int t    = threadIdx.x;
  const int lane = t & 31;
  const int wave = t >> 5;
  const int wm   = wave & 1;      // 0..1  (M sub-tile)
  const int wn   = wave >> 1;     // 0..3  (N sub-tile)
  const int nl   = lane & 15;     // M row (A) / N col (B) within 16
  const int kh   = lane >> 4;     // K half-select

  const int bm = blockIdx.y * BM;
  const int bn = blockIdx.x * BN;

  const float alpha = alpha_ptr ? *alpha_ptr : 0.0f;

  v8f acc[2][4] = {};             // 32(M) x 64(N) per wave

  const int am0    = (wm << 5) + nl;
  const int am1    = am0 + 16;
  const int half8  = kh << 3;     // A operand K base within group
  const int half16 = kh << 4;     // unused; B also uses half8

  // staging maps (per thread, per K-step)
  const int ar = t >> 2;                 // 0..63
  const int ac = (t & 3) << 3;           // 0,8,16,24 (halfwords)
  const int a_src = (bm + ar < M) ? (bm + ar) : (M - 1);

  for (int k0 = 0; k0 < HDIM; k0 += BK) {
    // ---- stage tiles: pure 16B copies (async global->LDS when available) ----
    cp16(Ahg + (size_t)a_src * HDIM + k0 + ac, &Ah[ar][ac]);
    cp16(Alg + (size_t)a_src * HDIM + k0 + ac, &Al[ar][ac]);
#pragma unroll
    for (int i = 0; i < 4; ++i) {
      const int idx = (t << 2) + i;
      const int r   = idx >> 2;          // 0..255
      const int c   = (idx & 3) << 3;
      cp16(Whg + (size_t)(bn + r) * HDIM + k0 + c, &Wh[r][c]);
      cp16(Wlg + (size_t)(bn + r) * HDIM + k0 + c, &Wl[r][c]);
    }
    cp_fence();
    __syncthreads();

    // ---- operands per ISA layouts ----
    // A 16x32: lane holds K = half8+0..7 and 16+half8+0..7
    const v16bf a_h0 = ld16(&Ah[am0][half8], &Ah[am0][16 + half8]);
    const v16bf a_l0 = ld16(&Al[am0][half8], &Al[am0][16 + half8]);
    const v16bf a_h1 = ld16(&Ah[am1][half8], &Ah[am1][16 + half8]);
    const v16bf a_l1 = ld16(&Al[am1][half8], &Al[am1][16 + half8]);
#pragma unroll
    for (int tt = 0; tt < 4; ++tt) {
      // B 32x16 (= W^T tile): lane holds K = half8+0..7 and 16+half8+0..7
      const int rw = (wn << 6) + (tt << 4) + nl;
      const v16bf b_h = ld16(&Wh[rw][half8], &Wh[rw][16 + half8]);
      const v16bf b_l = ld16(&Wl[rw][half8], &Wl[rw][16 + half8]);
      acc[0][tt] = __builtin_amdgcn_wmma_f32_16x16x32_bf16(
          false, a_h0, false, b_h, (short)0, acc[0][tt], false, false);
      acc[0][tt] = __builtin_amdgcn_wmma_f32_16x16x32_bf16(
          false, a_l0, false, b_h, (short)0, acc[0][tt], false, false);
      acc[0][tt] = __builtin_amdgcn_wmma_f32_16x16x32_bf16(
          false, a_h0, false, b_l, (short)0, acc[0][tt], false, false);
      acc[1][tt] = __builtin_amdgcn_wmma_f32_16x16x32_bf16(
          false, a_h1, false, b_h, (short)0, acc[1][tt], false, false);
      acc[1][tt] = __builtin_amdgcn_wmma_f32_16x16x32_bf16(
          false, a_l1, false, b_h, (short)0, acc[1][tt], false, false);
      acc[1][tt] = __builtin_amdgcn_wmma_f32_16x16x32_bf16(
          false, a_h1, false, b_l, (short)0, acc[1][tt], false, false);
    }
    __syncthreads();
  }

  // D layout: VGPR r -> M = r (lanes 0-15) / 8+r (lanes 16-31); lane&15 -> N
#pragma unroll
  for (int mi = 0; mi < 2; ++mi) {
    const int m0 = bm + (wm << 5) + (mi << 4) + (kh << 3);
#pragma unroll
    for (int tt = 0; tt < 4; ++tt) {
      const int n = bn + (wn << 6) + (tt << 4) + nl;
#pragma unroll
      for (int r = 0; r < 8; ++r) {
        const int m = m0 + r;
        if (m < M) {
          const size_t off = (size_t)m * HDIM + n;
          float val = acc[mi][tt][r];
          if (resid) val = resid[off] + alpha * val;
          C[off] = val;
        }
      }
    }
  }
}

// ---------------------------------------------------------------------------
// Fused scores -> softmax(S=10) -> context; ctx emitted as hi/lo bf16 planes.
// ---------------------------------------------------------------------------
__global__ __launch_bounds__(256)
void attn_fused(const float* __restrict__ q, const float* __restrict__ kbuf,
                const float* __restrict__ vbuf,
                ushort_t* __restrict__ ctxh, ushort_t* __restrict__ ctxl) {
  __shared__ float red[SSEQ][256];
  __shared__ float probs[SSEQ];

  const int row = blockIdx.x;           // 0 .. BL-1
  const int b   = row / LSEQ;
  const int t   = threadIdx.x;

  float qv[8];
#pragma unroll
  for (int j = 0; j < 8; ++j)
    qv[j] = q[(size_t)row * HDIM + t + (j << 8)];

  float part[SSEQ];
#pragma unroll
  for (int s = 0; s < SSEQ; ++s) {
    const float* kr = kbuf + (size_t)(b * SSEQ + s) * HDIM;
    float acc = 0.0f;
#pragma unroll
    for (int j = 0; j < 8; ++j) acc += qv[j] * kr[t + (j << 8)];
    part[s] = acc;
  }
#pragma unroll
  for (int s = 0; s < SSEQ; ++s) red[s][t] = part[s];
  __syncthreads();

  for (int off = 128; off > 0; off >>= 1) {
    if (t < off) {
#pragma unroll
      for (int s = 0; s < SSEQ; ++s) red[s][t] += red[s][t + off];
    }
    __syncthreads();
  }

  if (t == 0) {
    const float scale = 0.02209708691f;   // 1/sqrt(2048)
    float sc[SSEQ], mx = -3.4e38f;
#pragma unroll
    for (int s = 0; s < SSEQ; ++s) { sc[s] = red[s][0] * scale; mx = fmaxf(mx, sc[s]); }
    float sum = 0.0f;
#pragma unroll
    for (int s = 0; s < SSEQ; ++s) { sc[s] = __expf(sc[s] - mx); sum += sc[s]; }
    const float inv = 1.0f / sum;
#pragma unroll
    for (int s = 0; s < SSEQ; ++s) probs[s] = sc[s] * inv;
  }
  __syncthreads();

  float p[SSEQ];
#pragma unroll
  for (int s = 0; s < SSEQ; ++s) p[s] = probs[s];

#pragma unroll
  for (int j = 0; j < 8; ++j) {
    const int h = t + (j << 8);
    float c = 0.0f;
#pragma unroll
    for (int s = 0; s < SSEQ; ++s)
      c += p[s] * vbuf[(size_t)(b * SSEQ + s) * HDIM + h];
    ushort_t ch, cl;
    split_f32(c, ch, cl);
    ctxh[(size_t)row * HDIM + h] = ch;
    ctxl[(size_t)row * HDIM + h] = cl;
  }
}

// ---------------------------------------------------------------------------
extern "C" void kernel_launch(void* const* d_in, const int* in_sizes, int n_in,
                              void* d_out, int out_size, void* d_ws, size_t ws_size,
                              hipStream_t stream) {
  const float* h_eng = (const float*)d_in[0];  // [B, L, H]
  const float* h_loj = (const float*)d_in[1];  // [B, S, H]
  const float* w_q   = (const float*)d_in[2];  // [H, H]
  const float* w_k   = (const float*)d_in[3];
  const float* w_v   = (const float*)d_in[4];
  const float* w_o   = (const float*)d_in[5];
  const float* alpha = (const float*)d_in[6];  // scalar on device
  float*       out   = (float*)d_out;          // [B, L, H]

  char* p = (char*)d_ws;
  float* qbuf = (float*)p;      p += (size_t)BL * HDIM * 4;     // q (f32)
  ushort_t* heh = (ushort_t*)p; p += (size_t)BL * HDIM * 2;     // h_english hi
  ushort_t* hel = (ushort_t*)p; p += (size_t)BL * HDIM * 2;     // h_english lo
  ushort_t* cth = (ushort_t*)p; p += (size_t)BL * HDIM * 2;     // context hi
  ushort_t* ctl = (ushort_t*)p; p += (size_t)BL * HDIM * 2;     // context lo
  ushort_t* wqh = (ushort_t*)p; p += (size_t)HDIM * HDIM * 2;
  ushort_t* wql = (ushort_t*)p; p += (size_t)HDIM * HDIM * 2;
  ushort_t* wkh = (ushort_t*)p; p += (size_t)HDIM * HDIM * 2;
  ushort_t* wkl = (ushort_t*)p; p += (size_t)HDIM * HDIM * 2;
  ushort_t* wvh = (ushort_t*)p; p += (size_t)HDIM * HDIM * 2;
  ushort_t* wvl = (ushort_t*)p; p += (size_t)HDIM * HDIM * 2;
  ushort_t* woh = (ushort_t*)p; p += (size_t)HDIM * HDIM * 2;
  ushort_t* wol = (ushort_t*)p; p += (size_t)HDIM * HDIM * 2;
  ushort_t* ljh = (ushort_t*)p; p += (size_t)BS * HDIM * 2;
  ushort_t* ljl = (ushort_t*)p; p += (size_t)BS * HDIM * 2;
  float* kbuf = (float*)p;      p += (size_t)BS * HDIM * 4;
  float* vbuf = (float*)p;      p += (size_t)BS * HDIM * 4;

  const dim3 blk(256);
  const long nW  = (long)HDIM * HDIM / 4;   // float4 count per weight
  const long nE  = (long)BL * HDIM / 4;
  const long nLj = (long)BS * HDIM / 4;

  // one-shot f32 -> (hi,lo) bf16 plane splits
  split4_kernel<<<dim3(80),   blk, 0, stream>>>(h_loj, ljh, ljl, nLj);
  split4_kernel<<<dim3(4096), blk, 0, stream>>>(w_q, wqh, wql, nW);
  split4_kernel<<<dim3(4096), blk, 0, stream>>>(w_k, wkh, wkl, nW);
  split4_kernel<<<dim3(4096), blk, 0, stream>>>(w_v, wvh, wvl, nW);
  split4_kernel<<<dim3(4096), blk, 0, stream>>>(w_o, woh, wol, nW);
  split4_kernel<<<dim3(8192), blk, 0, stream>>>(h_eng, heh, hel, nE);

  const dim3 gSmall(HDIM / BN, 1);        // M=40 -> one 64-row tile
  const dim3 gBig(HDIM / BN, BL / BM);    // 8 x 256 blocks

  // k = h_lojban @ w_k^T ; v = h_lojban @ w_v^T
  gemm_bf16split_wmma<<<gSmall, blk, 0, stream>>>(ljh, ljl, wkh, wkl, kbuf, BS, nullptr, nullptr);
  gemm_bf16split_wmma<<<gSmall, blk, 0, stream>>>(ljh, ljl, wvh, wvl, vbuf, BS, nullptr, nullptr);
  // q = h_english @ w_q^T
  gemm_bf16split_wmma<<<gBig, blk, 0, stream>>>(heh, hel, wqh, wql, qbuf, BL, nullptr, nullptr);
  // scores -> softmax -> context (emitted as bf16 planes)
  attn_fused<<<dim3(BL), blk, 0, stream>>>(qbuf, kbuf, vbuf, cth, ctl);
  // out = h_english + alpha * (context @ w_o^T)
  gemm_bf16split_wmma<<<gBig, blk, 0, stream>>>(cth, ctl, woh, wol, out, BL, h_eng, alpha);
}
